// EncoderNet_37340445671621
// MI455X (gfx1250) — compile-verified, matrix operand order checked by reading, not run on hardware
//
#include <hip/hip_runtime.h>
#include <hip/hip_bf16.h>
#include <math.h>

// ---------------------------------------------------------------------------
// 3-layer GAT (num_heads=1) for MI455X (gfx1250, wave32).
//   per layer: h = x@W (WMMA bf16->f32), el/er = h.attn, edge softmax by dst,
//              out[dst] += a * h[src], + bias.
// All intermediate buffers (<=110MB) are L2-resident (192MB L2), so the
// random edge gathers/atomics hit L2, not HBM.
// ---------------------------------------------------------------------------

#define D 128
#define NEG_SLOPE 0.2f

typedef __bf16 bf16_t;
typedef bf16_t v16bf __attribute__((ext_vector_type(16)));
typedef float  v8f   __attribute__((ext_vector_type(8)));

// float -> bf16, round-to-nearest-even
__device__ __forceinline__ bf16_t f2bf(float f) {
    unsigned u = __float_as_uint(f);
    u += 0x7FFFu + ((u >> 16) & 1u);
    unsigned short h = (unsigned short)(u >> 16);
    return __builtin_bit_cast(bf16_t, h);
}

// order-preserving float<->uint map (for atomicMax on u32)
__device__ __forceinline__ unsigned fmapu(float f) {
    unsigned u = __float_as_uint(f);
    return (u & 0x80000000u) ? ~u : (u | 0x80000000u);
}
__device__ __forceinline__ float funmap(unsigned u) {
    return __uint_as_float((u & 0x80000000u) ? (u & 0x7FFFFFFFu) : ~u);
}

// ---------------------------------------------------------------------------
// Pack W [128x128 f32, row-major K x N] into the WMMA 16-bit B-operand layout.
// Lane L = n + 16*g (n = N-col within 16-tile, g = half select).
// elems j<8  -> K = 8g + j ;  elems j>=8 -> K = 16 + 8g + (j-8).
// Wp index: (((ntile*4 + kblock)*32 + lane)*16 + j), each lane's 16 bf16 are
// contiguous (32B) -> single wide load per WMMA in the GEMM.
// ---------------------------------------------------------------------------
__global__ void pack_w_kernel(const float* __restrict__ W, bf16_t* __restrict__ Wp) {
    int idx = blockIdx.x * blockDim.x + threadIdx.x;   // 8*4*32*16 = 16384
    if (idx >= 8 * 4 * 32 * 16) return;
    int j    = idx & 15;
    int lane = (idx >> 4) & 31;
    int kb   = (idx >> 9) & 3;
    int nt   = idx >> 11;
    int g = lane >> 4, n = lane & 15;
    int K = (j < 8) ? (8 * g + j) : (16 + 8 * g + (j - 8));
    Wp[idx] = f2bf(W[(size_t)(kb * 32 + K) * D + nt * 16 + n]);
}

// ---------------------------------------------------------------------------
// H[M x 128] = X[M x 128] @ W[128 x 128]   (bf16 WMMA, f32 accumulate)
// one wave per 16x16 output tile; block = 8 waves = 8 N-tiles of one M-tile.
// K loop fully unrolled -> 4 back-to-back v_wmma with loads hoisted.
// ---------------------------------------------------------------------------
__global__ void __launch_bounds__(256) gemm_wmma_kernel(
        const float* __restrict__ X, const bf16_t* __restrict__ Wp,
        float* __restrict__ H, int M) {
    int lane  = threadIdx.x & 31;
    int w     = threadIdx.x >> 5;     // ntile 0..7
    int mtile = blockIdx.x;
    int g = lane >> 4;
    int n = lane & 15;

    bool full = (mtile * 16 + 16) <= M;   // all rows of this tile valid?
    int row = mtile * 16 + n;             // A-operand row for this lane
    if (row >= M) row = M - 1;            // clamp for tail-tile loads

    v8f acc = {};
#pragma unroll
    for (int kb = 0; kb < 4; ++kb) {
        int k0 = kb * 32;
        // A 16x32 bf16 operand: elems 0..7 = K(8g..8g+7), 8..15 = K(+16)
        const float* xr = X + (size_t)row * D + k0 + 8 * g;
        float4 p0 = *(const float4*)(xr);
        float4 p1 = *(const float4*)(xr + 4);
        float4 q0 = *(const float4*)(xr + 16);
        float4 q1 = *(const float4*)(xr + 20);
        v16bf a;
        a[0]  = f2bf(p0.x); a[1]  = f2bf(p0.y); a[2]  = f2bf(p0.z); a[3]  = f2bf(p0.w);
        a[4]  = f2bf(p1.x); a[5]  = f2bf(p1.y); a[6]  = f2bf(p1.z); a[7]  = f2bf(p1.w);
        a[8]  = f2bf(q0.x); a[9]  = f2bf(q0.y); a[10] = f2bf(q0.z); a[11] = f2bf(q0.w);
        a[12] = f2bf(q1.x); a[13] = f2bf(q1.y); a[14] = f2bf(q1.z); a[15] = f2bf(q1.w);

        // pre-packed B operand: one contiguous 32B load per lane
        v16bf b = *(const v16bf*)(Wp + (((size_t)w * 4 + kb) * 32 + lane) * 16);

        acc = __builtin_amdgcn_wmma_f32_16x16x32_bf16(
                  false, a, false, b, (short)0, acc, false, false);
    }

    // D layout: VGPR r, lanes 0-15 -> M=r ; lanes 16-31 -> M=8+r ; N=lane&15
    int mbase = mtile * 16 + 8 * g;
    float* hp = H + (size_t)mbase * D + w * 16 + n;
    if (full) {
        // fast path: straight-line stores, no exec-mask branching
#pragma unroll
        for (int r = 0; r < 8; ++r) hp[(size_t)r * D] = acc[r];
    } else {
#pragma unroll
        for (int r = 0; r < 8; ++r)
            if (mbase + r < M) hp[(size_t)r * D] = acc[r];
    }
}

// ---------------------------------------------------------------------------
// el[n] = h[n,:].attn_l ; er[n] = h[n,:].attn_r   (one wave per node)
// ---------------------------------------------------------------------------
__global__ void elr_kernel(const float* __restrict__ H,
                           const float* __restrict__ al, const float* __restrict__ ar,
                           float* __restrict__ el, float* __restrict__ er, int N) {
    int t = blockIdx.x * blockDim.x + threadIdx.x;
    int node = t >> 5, lane = t & 31;
    if (node >= N) return;
    float4 hv = *(const float4*)(H + (size_t)node * D + lane * 4);
    float4 av = *(const float4*)(al + lane * 4);
    float4 bv = *(const float4*)(ar + lane * 4);
    float pl = hv.x * av.x + hv.y * av.y + hv.z * av.z + hv.w * av.w;
    float pr = hv.x * bv.x + hv.y * bv.y + hv.z * bv.z + hv.w * bv.w;
#pragma unroll
    for (int off = 16; off; off >>= 1) {
        pl += __shfl_down(pl, off, 32);
        pr += __shfl_down(pr, off, 32);
    }
    if (lane == 0) { el[node] = pl; er[node] = pr; }
}

// out <- bias broadcast; m_u <- 0 (mapped, below every real); s <- 0
__global__ void init_kernel(float* __restrict__ out, const float* __restrict__ bias,
                            unsigned* __restrict__ mu, float* __restrict__ ssum,
                            int N) {
    size_t idx = (size_t)blockIdx.x * blockDim.x + threadIdx.x;
    if (idx >= (size_t)N * D) return;
    out[idx] = bias[idx & (D - 1)];
    if ((idx & (D - 1)) == 0) { mu[idx >> 7] = 0u; ssum[idx >> 7] = 0.0f; }
}

// e = leaky_relu(el[src]+er[dst]); segment max by dst via u32 atomicMax
__global__ void edge_max_kernel(const int* __restrict__ src, const int* __restrict__ dst,
                                const float* __restrict__ el, const float* __restrict__ er,
                                float* __restrict__ ebuf, unsigned* __restrict__ mu, int E) {
    int i = blockIdx.x * blockDim.x + threadIdx.x;
    if (i >= E) return;
    float v = el[src[i]] + er[dst[i]];
    v = (v > 0.0f) ? v : NEG_SLOPE * v;
    ebuf[i] = v;
    atomicMax(&mu[dst[i]], fmapu(v));
}

// ex = exp(e - m[dst]); segment sum by dst
__global__ void edge_sum_kernel(const int* __restrict__ dst, float* __restrict__ ebuf,
                                const unsigned* __restrict__ mu, float* __restrict__ ssum,
                                int E) {
    int i = blockIdx.x * blockDim.x + threadIdx.x;
    if (i >= E) return;
    int d  = dst[i];
    float ex = expf(ebuf[i] - funmap(mu[d]));
    ebuf[i] = ex;
    atomicAdd(&ssum[d], ex);
}

// out[dst,:] += (ex/s[dst]) * h[src,:]   (one wave per edge, 4 floats/lane)
__global__ void __launch_bounds__(256) edge_aggr_kernel(
        const int* __restrict__ src, const int* __restrict__ dst,
        const float* __restrict__ ebuf, const float* __restrict__ ssum,
        const float* __restrict__ H, float* __restrict__ out, int E) {
    int t = blockIdx.x * blockDim.x + threadIdx.x;
    int e = t >> 5, lane = t & 31;
    if (e >= E) return;
    int sN = src[e], dN = dst[e];
    float a = ebuf[e] / ssum[dN];
    float4 hv = *(const float4*)(H + (size_t)sN * D + lane * 4);
    float* op = out + (size_t)dN * D + lane * 4;
    atomicAdd(op + 0, a * hv.x);
    atomicAdd(op + 1, a * hv.y);
    atomicAdd(op + 2, a * hv.z);
    atomicAdd(op + 3, a * hv.w);
}

// ---------------------------------------------------------------------------
extern "C" void kernel_launch(void* const* d_in, const int* in_sizes, int n_in,
                              void* d_out, int out_size, void* d_ws, size_t ws_size,
                              hipStream_t stream) {
    const float* feat = (const float*)d_in[0];
    const int*   src  = (const int*)d_in[1];
    const int*   dst  = (const int*)d_in[2];
    const float* Wl[3]  = {(const float*)d_in[3],  (const float*)d_in[7],  (const float*)d_in[11]};
    const float* ALl[3] = {(const float*)d_in[4],  (const float*)d_in[8],  (const float*)d_in[12]};
    const float* ARl[3] = {(const float*)d_in[5],  (const float*)d_in[9],  (const float*)d_in[13]};
    const float* Bl[3]  = {(const float*)d_in[6],  (const float*)d_in[10], (const float*)d_in[14]};

    const int N = in_sizes[0] / D;
    const int E = in_sizes[1];

    // workspace carve-up (256B aligned slices)
    char* ws = (char*)d_ws;
    size_t off = 0;
    auto carve = [&](size_t bytes) -> void* {
        void* p = ws + off;
        off = (off + bytes + 255) & ~(size_t)255;
        return p;
    };
    float*    Hbuf = (float*)   carve((size_t)N * D * sizeof(float));  // 51.2 MB
    float*    Abuf = (float*)   carve((size_t)N * D * sizeof(float));  // 51.2 MB
    float*    el   = (float*)   carve((size_t)N * sizeof(float));
    float*    er   = (float*)   carve((size_t)N * sizeof(float));
    float*    ssum = (float*)   carve((size_t)N * sizeof(float));
    unsigned* mu   = (unsigned*)carve((size_t)N * sizeof(unsigned));
    float*    ebuf = (float*)   carve((size_t)E * sizeof(float));      // 6.4 MB
    bf16_t*   Wp   = (bf16_t*)  carve((size_t)D * D * sizeof(bf16_t)); // 32 KB

    const int blk = 256;
    const int gPack = (8 * 4 * 32 * 16 + blk - 1) / blk;
    const int gGemm = (N + 15) / 16;                              // 6250
    const int gElr  = (int)(((size_t)N * 32 + blk - 1) / blk);
    const int gInit = (int)(((size_t)N * D + blk - 1) / blk);
    const int gEdge = (E + blk - 1) / blk;
    const int gAggr = (int)(((size_t)E * 32 + blk - 1) / blk);

    const float* X = feat;
    for (int L = 0; L < 3; ++L) {
        float* O = (L == 2) ? (float*)d_out : Abuf;

        pack_w_kernel   <<<gPack, blk, 0, stream>>>(Wl[L], Wp);
        gemm_wmma_kernel<<<gGemm, blk, 0, stream>>>(X, Wp, Hbuf, N);
        elr_kernel      <<<gElr,  blk, 0, stream>>>(Hbuf, ALl[L], ARl[L], el, er, N);
        init_kernel     <<<gInit, blk, 0, stream>>>(O, Bl[L], mu, ssum, N);
        edge_max_kernel <<<gEdge, blk, 0, stream>>>(src, dst, el, er, ebuf, mu, E);
        edge_sum_kernel <<<gEdge, blk, 0, stream>>>(dst, ebuf, mu, ssum, E);
        edge_aggr_kernel<<<gAggr, blk, 0, stream>>>(src, dst, ebuf, ssum, Hbuf, O, E);

        X = O;   // layer output feeds next layer (GEMM reads it before init_kernel rewrites)
    }
}
